// Quantized_10385230922551
// MI455X (gfx1250) — matrix operand order repaired and checked by reading, not compiled, not run on hardware
//
#include <hip/hip_runtime.h>

// CDNA5 / gfx1250 fused dequant-GEMM:
//   out[m,n] = sum_k (y[m,k]*rng[k]) * W[k,n]  +  c[m],   c[m] = sum_k y[m,k]*off[k]
// M = B*T = 4096, N = 4096, K = C = 4096.
//
// f16 WMMA (v_wmma_f32_16x16x32_f16), f32 accumulation. Weight codes (0..255)
// are exact in f16; scale applied to A in f32 before conversion; offset
// correction kept entirely in f32 (VALU + ds_add_f32 reduction).
//
// Pipeline: 2x-unrolled ping/pong register double-buffer; tile t+1's global
// loads issue between the LDS-store barrier and the WMMA block of tile t.

typedef _Float16 v2h  __attribute__((ext_vector_type(2)));
typedef _Float16 v4h  __attribute__((ext_vector_type(4)));
typedef _Float16 v8h  __attribute__((ext_vector_type(8)));
typedef _Float16 v16h __attribute__((ext_vector_type(16)));
typedef float    v8f  __attribute__((ext_vector_type(8)));

#define BM 128
#define BN 128
#define BK 32
#define LDA 40   // padded f16 row stride for As[BM][LDA]  (conflict-free b128 frag loads)
#define LDB 40   // padded f16 row stride for Bs[BN][LDB]  (B transposed: Bs[n][k])

union U16 { v16h v; v8h h[2]; };

__global__ __launch_bounds__(256)
void dequant_gemm_wmma(const float* __restrict__ y,
                       const float* __restrict__ w,
                       const float* __restrict__ rng,
                       const float* __restrict__ off,
                       float* __restrict__ out,
                       int M, int N, int K)
{
    __shared__ _Float16 As[BM][LDA];
    __shared__ _Float16 Bs[BN][LDB];
    __shared__ float    Cfix[BM];

    const int tid     = threadIdx.x;
    const int lane    = tid & 31;          // wave32
    const int wave    = tid >> 5;          // 8 waves
    const int wm      = wave & 1;          // M half (64 rows)
    const int wn      = wave >> 1;         // N quarter (32 cols)
    const int m0      = blockIdx.y * BM;
    const int n0      = blockIdx.x * BN;
    const int lane_lo = lane & 15;
    const int kh      = lane >> 4;

    const int a_cg = (tid & 7) * 4;        // A: k-subgroup of 4
    const int a_rb = tid >> 3;             // A: row base 0..31
    const int b_n  = tid & 127;            // B: column
    const int b_kp = tid >> 7;             // B: k-pair group (wave-uniform)

    if (tid < BM) Cfix[tid] = 0.0f;

    v8f acc[4][2];
    #pragma unroll
    for (int r = 0; r < 4; ++r)
        #pragma unroll
        for (int c = 0; c < 2; ++c)
            #pragma unroll
            for (int i = 0; i < 8; ++i) acc[r][c][i] = 0.0f;

    float cacc[4] = {0.0f, 0.0f, 0.0f, 0.0f};

    // ping/pong register-staged tiles
    float4 t0_a[4], t1_a[4];
    float  t0_w[16], t1_w[16];
    float4 t0_rng, t0_off, t1_rng, t1_off;

    const float* yA = y + (size_t)m0 * K + a_cg;   // + row*K + k0
    const float* wB = w + n0 + b_n;                // + k*N

#define LOAD_TILE(T_, K0_)                                                    \
    {                                                                         \
        T_##_rng = *(const float4*)(rng + (K0_) + a_cg);                      \
        T_##_off = *(const float4*)(off + (K0_) + a_cg);                      \
        _Pragma("unroll")                                                     \
        for (int p = 0; p < 4; ++p)                                           \
            T_##_a[p] = *(const float4*)(yA + (size_t)(a_rb + 32 * p) * K + (K0_)); \
        _Pragma("unroll")                                                     \
        for (int p = 0; p < 8; ++p) {                                         \
            const int k = (2 * p + b_kp) * 2;                                 \
            T_##_w[2 * p]     = wB[(size_t)((K0_) + k) * N];                  \
            T_##_w[2 * p + 1] = wB[(size_t)((K0_) + k + 1) * N];              \
        }                                                                     \
    }

#define GEMM_STEP(C_, N_, K0NEXT_, DOLOAD_)                                   \
    {                                                                         \
        __syncthreads();                                                      \
        /* store staged tile to LDS (f32->f16), accumulate correction */      \
        _Pragma("unroll")                                                     \
        for (int p = 0; p < 4; ++p) {                                         \
            const float4 v = C_##_a[p];                                       \
            v4h h = { (_Float16)(v.x * C_##_rng.x), (_Float16)(v.y * C_##_rng.y), \
                      (_Float16)(v.z * C_##_rng.z), (_Float16)(v.w * C_##_rng.w) }; \
            *(v4h*)(&As[a_rb + 32 * p][a_cg]) = h;                            \
            cacc[p] += v.x * C_##_off.x + v.y * C_##_off.y                    \
                     + v.z * C_##_off.z + v.w * C_##_off.w;                   \
        }                                                                     \
        _Pragma("unroll")                                                     \
        for (int p = 0; p < 8; ++p) {                                         \
            const int k = (2 * p + b_kp) * 2;                                 \
            v2h d = { (_Float16)C_##_w[2 * p], (_Float16)C_##_w[2 * p + 1] }; \
            *(v2h*)(&Bs[b_n][k]) = d;                                         \
        }                                                                     \
        __syncthreads();                                                      \
        /* issue next tile's global loads; wait lands after the WMMAs */      \
        if (DOLOAD_) LOAD_TILE(N_, K0NEXT_);                                  \
        /* fragment loads per ISA VGPR layouts */                             \
        v16h afr[4];                                                          \
        _Pragma("unroll")                                                     \
        for (int r = 0; r < 4; ++r) {                                         \
            const int m = wm * 64 + r * 16 + lane_lo;                         \
            U16 u;                                                            \
            u.h[0] = *(const v8h*)(&As[m][kh * 8]);                           \
            u.h[1] = *(const v8h*)(&As[m][16 + kh * 8]);                      \
            afr[r] = u.v;                                                     \
        }                                                                     \
        v16h bfr[2];                                                          \
        _Pragma("unroll")                                                     \
        for (int c = 0; c < 2; ++c) {                                         \
            const int nn = wn * 32 + c * 16 + lane_lo;                        \
            U16 u;                                                            \
            u.h[0] = *(const v8h*)(&Bs[nn][kh * 8]);                          \
            u.h[1] = *(const v8h*)(&Bs[nn][16 + kh * 8]);                     \
            bfr[c] = u.v;                                                     \
        }                                                                     \
        _Pragma("unroll")                                                     \
        for (int r = 0; r < 4; ++r)                                           \
            _Pragma("unroll")                                                 \
            for (int c = 0; c < 2; ++c)                                       \
                acc[r][c] = __builtin_amdgcn_wmma_f32_16x16x32_f16(           \
                    false, afr[r], false, bfr[c],                             \
                    (short)0, acc[r][c], false, false);                       \
    }

    const int ntiles = K / BK;
    LOAD_TILE(t0, 0);

    int t = 0;
    for (; t + 2 <= ntiles; t += 2) {
        GEMM_STEP(t0, t1, (t + 1) * BK, (t + 1) < ntiles);
        GEMM_STEP(t1, t0, (t + 2) * BK, (t + 2) < ntiles);
    }
    if (t < ntiles) {          // odd tail (not hit for K=4096)
        GEMM_STEP(t0, t1, 0, false);
    }

#undef GEMM_STEP
#undef LOAD_TILE

    // ---- reduce per-row correction (8 k-slice threads per row) ----
    #pragma unroll
    for (int p = 0; p < 4; ++p)
        atomicAdd(&Cfix[a_rb + 32 * p], cacc[p]);
    __syncthreads();

    // ---- epilogue: C/D layout (VGPR i -> M = kh*8 + i, N = lane_lo) + correction ----
    #pragma unroll
    for (int r = 0; r < 4; ++r) {
        const v8f cf = *(const v8f*)(&Cfix[wm * 64 + r * 16 + kh * 8]);
        #pragma unroll
        for (int c = 0; c < 2; ++c) {
            const int nn = n0 + wn * 32 + c * 16 + lane_lo;
            #pragma unroll
            for (int i = 0; i < 8; ++i) {
                const int m = m0 + wm * 64 + r * 16 + kh * 8 + i;
                out[(size_t)m * N + nn] = acc[r][c][i] + cf[i];
            }
        }
    }
}

extern "C" void kernel_launch(void* const* d_in, const int* in_sizes, int n_in,
                              void* d_out, int out_size, void* d_ws, size_t ws_size,
                              hipStream_t stream) {
    const float* y   = (const float*)d_in[0];   // [B,T,C] f32
    const float* w   = (const float*)d_in[1];   // [C,N]   f32 (uint8 codes as float)
    const float* rng = (const float*)d_in[2];   // [1,1,C] f32
    const float* off = (const float*)d_in[3];   // [C]     f32
    float* out = (float*)d_out;                 // [B,T,N] f32

    const int C = in_sizes[3];          // offset length = C
    const int N = in_sizes[1] / C;      // weight is C*N
    const int M = in_sizes[0] / C;      // y is (B*T)*C

    dim3 grid(N / BN, M / BM, 1);
    dequant_gemm_wmma<<<grid, dim3(256, 1, 1), 0, stream>>>(y, w, rng, off, out, M, N, C);
}
